// LSTMSampler_26439818674724
// MI455X (gfx1250) — matrix operand order repaired and checked by reading, not compiled, not run on hardware
//
#include <hip/hip_runtime.h>

// ---------------- problem constants (match reference) ----------------
#define BSZ     512
#define SRCLEN  128
#define ENCD    512
#define HDIM    1024
#define ZDIM    128
#define SEQL    64
#define H4      4096            // 4*HDIM
#define EPS_STD 1e-4f

// ---------------- WMMA types / helpers ----------------
typedef __attribute__((ext_vector_type(16))) __bf16 v16bf;
typedef __attribute__((ext_vector_type(8)))  float  v8f;

__device__ __forceinline__ unsigned short f2bf(float x) {
  union { float f; unsigned u; } v; v.f = x;
  unsigned r = v.u + 0x7FFFu + ((v.u >> 16) & 1u);   // round-to-nearest-even
  return (unsigned short)(r >> 16);
}

// A fragment: 16x32 bf16 tile of row-major [M,K] matrix (lda in elements).
// CDNA5 layout: lane m = lane&15, group g = lane>>4;
//   elems 0..7  -> K = 8g + j ; elems 8..15 -> K = 16 + 8g + j
__device__ __forceinline__ v16bf frag_a(const unsigned short* p, int lda, int lane) {
  const unsigned short* q = p + (size_t)(lane & 15) * lda + ((lane >> 4) << 3);
  union { v16bf v; uint4 u[2]; } r;
  r.u[0] = *(const uint4*)q;
  r.u[1] = *(const uint4*)(q + 16);
  return r.v;
}

// B fragment: 32x16 bf16. B(k,n) = W[n*ldw + k] (W row-major [N,K], i.e. x@W^T).
// CDNA5 layout: lane n = lane&15, group g = lane>>4;
//   elems 0..7  -> K = 8g + j ; elems 8..15 -> K = 16 + 8g + j
__device__ __forceinline__ v16bf frag_b(const unsigned short* p, int ldw, int lane) {
  const unsigned short* q = p + (size_t)(lane & 15) * ldw + ((lane >> 4) << 3);
  union { v16bf v; uint4 u[2]; } r;
  r.u[0] = *(const uint4*)q;
  r.u[1] = *(const uint4*)(q + 16);
  return r.v;
}

#define WMMA_BF16(a, b, c) \
  __builtin_amdgcn_wmma_f32_16x16x32_bf16(false, (a), false, (b), (short)0, (c), false, false)

// ---------------- weight packing (f32 -> bf16 bits) ----------------
__global__ void k_pack_bf16(const float* __restrict__ src, unsigned short* __restrict__ dst,
                            int rows, int cols, int dld) {
  int i = blockIdx.x * blockDim.x + threadIdx.x;
  if (i >= rows * cols) return;
  int r = i / cols, c = i - r * cols;
  dst[(size_t)r * dld + c] = f2bf(src[i]);
}

__global__ void k_addf(const float* __restrict__ a, const float* __restrict__ b,
                       float* __restrict__ d, int n) {
  int i = blockIdx.x * blockDim.x + threadIdx.x;
  if (i < n) d[i] = a[i] + b[i];
}

// ---------------- state init: h0=h1=c0=c1=0, z = broadcast(init_z) ----------------
__global__ void k_init_state(const float* __restrict__ initz,
                             unsigned short* __restrict__ X0,  // [BSZ][ZDIM+HDIM]
                             unsigned short* __restrict__ X1,  // [BSZ][2*HDIM]
                             unsigned short* __restrict__ XA,  // [BSZ][ENCD+HDIM]
                             float* __restrict__ c0, float* __restrict__ c1) {
  int i = blockIdx.x * blockDim.x + threadIdx.x;
  if (i >= BSZ * HDIM) return;
  int b = i >> 10, j = i & (HDIM - 1);
  c0[i] = 0.f; c1[i] = 0.f;
  X0[(size_t)b * (ZDIM + HDIM) + ZDIM + j] = 0;   // h0
  X1[(size_t)b * (2 * HDIM) + j]          = 0;    // h0
  X1[(size_t)b * (2 * HDIM) + HDIM + j]   = 0;    // h1
  XA[(size_t)b * (ENCD + HDIM) + ENCD + j] = 0;   // h1
  if (j < ZDIM) X0[(size_t)b * (ZDIM + HDIM) + j] = f2bf(initz[j]);
}

// ---------------- GEMM: C_f32 = A_bf16 @ W_bf16^T + bias ; wave = 32x64 tile ----------------
__global__ __launch_bounds__(256) void k_gemm_bias_f32(
    const unsigned short* __restrict__ A, int lda,
    const unsigned short* __restrict__ W, int ldw,
    const float* __restrict__ bias,
    float* __restrict__ C, int ldc, int M, int N, int K) {
  int lane = threadIdx.x & 31;
  int wave = (int)((blockIdx.x * blockDim.x + threadIdx.x) >> 5);
  int ntn = N >> 6;
  int mb = wave / ntn, nb = wave - mb * ntn;
  int m0 = mb << 5, n0 = nb << 6;
  if (m0 >= M) return;
  v8f acc[2][4] = {};
  for (int k = 0; k < K; k += 32) {
    v16bf a0 = frag_a(A + (size_t)m0 * lda + k, lda, lane);
    v16bf a1 = frag_a(A + (size_t)(m0 + 16) * lda + k, lda, lane);
#pragma unroll
    for (int ni = 0; ni < 4; ++ni) {
      v16bf bw = frag_b(W + (size_t)(n0 + 16 * ni) * ldw + k, ldw, lane);
      acc[0][ni] = WMMA_BF16(a0, bw, acc[0][ni]);
      acc[1][ni] = WMMA_BF16(a1, bw, acc[1][ni]);
    }
  }
  int cn = lane & 15, mo = (lane >> 4) << 3;
#pragma unroll
  for (int mi = 0; mi < 2; ++mi)
#pragma unroll
    for (int ni = 0; ni < 4; ++ni) {
      int nn = n0 + 16 * ni + cn;
      float bb = bias ? bias[nn] : 0.f;
      float* cp = C + (size_t)(m0 + 16 * mi + mo) * ldc + nn;
#pragma unroll
      for (int r = 0; r < 8; ++r) cp[(size_t)r * ldc] = acc[mi][ni][r] + bb;
    }
}

// ---------------- GEMM: C_bf16 = tanh(A @ W^T) (attn out-proj) ----------------
__global__ __launch_bounds__(256) void k_gemm_tanh_bf16(
    const unsigned short* __restrict__ A, int lda,
    const unsigned short* __restrict__ W, int ldw,
    unsigned short* __restrict__ C, int ldc, int M, int N, int K) {
  int lane = threadIdx.x & 31;
  int wave = (int)((blockIdx.x * blockDim.x + threadIdx.x) >> 5);
  int ntn = N >> 6;
  int mb = wave / ntn, nb = wave - mb * ntn;
  int m0 = mb << 5, n0 = nb << 6;
  if (m0 >= M) return;
  v8f acc[2][4] = {};
  for (int k = 0; k < K; k += 32) {
    v16bf a0 = frag_a(A + (size_t)m0 * lda + k, lda, lane);
    v16bf a1 = frag_a(A + (size_t)(m0 + 16) * lda + k, lda, lane);
#pragma unroll
    for (int ni = 0; ni < 4; ++ni) {
      v16bf bw = frag_b(W + (size_t)(n0 + 16 * ni) * ldw + k, ldw, lane);
      acc[0][ni] = WMMA_BF16(a0, bw, acc[0][ni]);
      acc[1][ni] = WMMA_BF16(a1, bw, acc[1][ni]);
    }
  }
  int cn = lane & 15, mo = (lane >> 4) << 3;
#pragma unroll
  for (int mi = 0; mi < 2; ++mi)
#pragma unroll
    for (int ni = 0; ni < 4; ++ni) {
      int nn = n0 + 16 * ni + cn;
      unsigned short* cp = C + (size_t)(m0 + 16 * mi + mo) * ldc + nn;
#pragma unroll
      for (int r = 0; r < 8; ++r) cp[(size_t)r * ldc] = f2bf(tanhf(acc[mi][ni][r]));
    }
}

// ---- fused mu/sigma GEMM + reparameterization sample; writes z (bf16) for next step ----
__global__ __launch_bounds__(256) void k_gemm_musg_sample(
    const unsigned short* __restrict__ CXT,   // [BSZ][HDIM] bf16
    const unsigned short* __restrict__ Wm,    // [ZDIM][HDIM] bf16
    const unsigned short* __restrict__ Ws,    // [ZDIM][HDIM] bf16
    const float* __restrict__ mub, const float* __restrict__ sgb,
    const float* __restrict__ eps_t,          // [BSZ][ZDIM] f32
    unsigned short* __restrict__ X0z,         // z cols of X0, ld ZDIM+HDIM
    unsigned short* __restrict__ Zt) {        // Zall slice, ld ZDIM
  int lane = threadIdx.x & 31;
  int wave = (int)((blockIdx.x * blockDim.x + threadIdx.x) >> 5);
  int mb = wave >> 3, nb = wave & 7;          // 16 x 8 wave tiles
  int m0 = mb << 5, n0 = nb << 4;
  v8f am[2] = {}, as_[2] = {};
  for (int k = 0; k < HDIM; k += 32) {
    v16bf a0 = frag_a(CXT + (size_t)m0 * HDIM + k, HDIM, lane);
    v16bf a1 = frag_a(CXT + (size_t)(m0 + 16) * HDIM + k, HDIM, lane);
    v16bf bm = frag_b(Wm + (size_t)n0 * HDIM + k, HDIM, lane);
    v16bf bs = frag_b(Ws + (size_t)n0 * HDIM + k, HDIM, lane);
    am[0] = WMMA_BF16(a0, bm, am[0]);  am[1] = WMMA_BF16(a1, bm, am[1]);
    as_[0] = WMMA_BF16(a0, bs, as_[0]); as_[1] = WMMA_BF16(a1, bs, as_[1]);
  }
  int cn = lane & 15, mo = (lane >> 4) << 3;
  int nn = n0 + cn;
  float bm_ = mub[nn], bs_ = sgb[nn];
#pragma unroll
  for (int mi = 0; mi < 2; ++mi)
#pragma unroll
    for (int r = 0; r < 8; ++r) {
      int mm = m0 + 16 * mi + mo + r;
      float mu = am[mi][r] + bm_;
      float sg = as_[mi][r] + bs_;
      float z = mu + eps_t[(size_t)mm * ZDIM + nn] * EPS_STD * expf(-0.5f * fabsf(sg));
      unsigned short zb = f2bf(z);
      X0z[(size_t)mm * (ZDIM + HDIM) + nn] = zb;
      Zt[(size_t)mm * ZDIM + nn] = zb;
    }
}

// ---------------- final out = Z @ fcW^T + fcb, scattered as [b][t][e] ----------------
__global__ __launch_bounds__(256) void k_gemm_fc(
    const unsigned short* __restrict__ Z,     // [(t*BSZ+b)][ZDIM] bf16
    const unsigned short* __restrict__ Wfc,   // [ENCD][ZDIM] bf16
    const float* __restrict__ fcb, float* __restrict__ out) {
  int lane = threadIdx.x & 31;
  int wave = (int)((blockIdx.x * blockDim.x + threadIdx.x) >> 5);
  int nb = wave & 7, mb = wave >> 3;          // ENCD/64 = 8 n-blocks
  int m0 = mb << 5, n0 = nb << 6;
  v8f acc[2][4] = {};
  for (int k = 0; k < ZDIM; k += 32) {
    v16bf a0 = frag_a(Z + (size_t)m0 * ZDIM + k, ZDIM, lane);
    v16bf a1 = frag_a(Z + (size_t)(m0 + 16) * ZDIM + k, ZDIM, lane);
#pragma unroll
    for (int ni = 0; ni < 4; ++ni) {
      v16bf bw = frag_b(Wfc + (size_t)(n0 + 16 * ni) * ZDIM + k, ZDIM, lane);
      acc[0][ni] = WMMA_BF16(a0, bw, acc[0][ni]);
      acc[1][ni] = WMMA_BF16(a1, bw, acc[1][ni]);
    }
  }
  int cn = lane & 15, mo = (lane >> 4) << 3;
#pragma unroll
  for (int mi = 0; mi < 2; ++mi)
#pragma unroll
    for (int ni = 0; ni < 4; ++ni) {
      int nn = n0 + 16 * ni + cn;
      float bb = fcb[nn];
#pragma unroll
      for (int r = 0; r < 8; ++r) {
        int row = m0 + 16 * mi + mo + r;          // row = t*BSZ + b
        int tt = row >> 9, b = row & (BSZ - 1);
        out[(size_t)b * (SEQL * ENCD) + (size_t)tt * ENCD + nn] = acc[mi][ni][r] + bb;
      }
    }
}

// ---------------- LSTM cell pointwise: gates -> (c, h) ; h written to 2 bf16 slots ----------------
__global__ __launch_bounds__(256) void k_lstm_cell(
    const float* __restrict__ gates, float* __restrict__ c,
    unsigned short* __restrict__ h0d, int ld0,
    unsigned short* __restrict__ h1d, int ld1) {
  int i = blockIdx.x * blockDim.x + threadIdx.x;
  if (i >= BSZ * HDIM) return;
  int b = i >> 10, j = i & (HDIM - 1);
  const float* g = gates + (size_t)b * H4;
  float ig = 1.f / (1.f + expf(-g[j]));
  float fg = 1.f / (1.f + expf(-g[HDIM + j]));
  float gg = tanhf(g[2 * HDIM + j]);
  float og = 1.f / (1.f + expf(-g[3 * HDIM + j]));
  float cn = fg * c[i] + ig * gg;
  c[i] = cn;
  unsigned short hb = f2bf(og * tanhf(cn));
  h0d[(size_t)b * ld0 + j] = hb;
  h1d[(size_t)b * ld1 + j] = hb;
}

// ---------------- attention: scores -> softmax -> context (bf16 into XA) ----------------
__global__ __launch_bounds__(128) void k_attn(
    const float* __restrict__ enc,            // [BSZ][SRCLEN][ENCD] f32
    const float* __restrict__ q,              // [BSZ][ENCD] f32
    unsigned short* __restrict__ ctx_dst, int ldc) {
  __shared__ float qs[ENCD];
  __shared__ float red[SRCLEN];
  __shared__ float attnv[SRCLEN];
  int b = blockIdx.x, t = threadIdx.x;
  const float* eb = enc + (size_t)b * SRCLEN * ENCD;
  for (int e = t; e < ENCD; e += 128) qs[e] = q[(size_t)b * ENCD + e];
  __syncthreads();
  const float* row = eb + (size_t)t * ENCD;
  float sc = 0.f;
  for (int e = 0; e < ENCD; e += 4) {
    float4 v = *(const float4*)(row + e);
    sc += v.x * qs[e] + v.y * qs[e + 1] + v.z * qs[e + 2] + v.w * qs[e + 3];
  }
  red[t] = sc; __syncthreads();
  for (int o = 64; o > 0; o >>= 1) { if (t < o) red[t] = fmaxf(red[t], red[t + o]); __syncthreads(); }
  float mx = red[0]; __syncthreads();
  float ex = expf(sc - mx);
  red[t] = ex; __syncthreads();
  for (int o = 64; o > 0; o >>= 1) { if (t < o) red[t] += red[t + o]; __syncthreads(); }
  attnv[t] = ex / red[0];
  __syncthreads();
  float a0 = 0.f, a1 = 0.f, a2 = 0.f, a3 = 0.f;
  for (int s = 0; s < SRCLEN; ++s) {
    float a = attnv[s];
    const float* r2 = eb + (size_t)s * ENCD + t;
    a0 += a * r2[0]; a1 += a * r2[128]; a2 += a * r2[256]; a3 += a * r2[384];
  }
  unsigned short* cd = ctx_dst + (size_t)b * ldc + t;
  cd[0] = f2bf(a0); cd[128] = f2bf(a1); cd[256] = f2bf(a2); cd[384] = f2bf(a3);
}

// ---------------- host orchestration ----------------
extern "C" void kernel_launch(void* const* d_in, const int* in_sizes, int n_in,
                              void* d_out, int out_size, void* d_ws, size_t ws_size,
                              hipStream_t stream) {
  (void)in_sizes; (void)n_in; (void)out_size; (void)ws_size;
  const float* enc   = (const float*)d_in[0];
  const float* eps   = (const float*)d_in[1];
  const float* initz = (const float*)d_in[2];
  const float* Wih0  = (const float*)d_in[3];
  const float* Whh0  = (const float*)d_in[4];
  const float* bih0  = (const float*)d_in[5];
  const float* bhh0  = (const float*)d_in[6];
  const float* Wih1  = (const float*)d_in[7];
  const float* Whh1  = (const float*)d_in[8];
  const float* bih1  = (const float*)d_in[9];
  const float* bhh1  = (const float*)d_in[10];
  const float* Win   = (const float*)d_in[11];
  const float* Wout  = (const float*)d_in[12];
  const float* muW   = (const float*)d_in[13];
  const float* mub   = (const float*)d_in[14];
  const float* sgW   = (const float*)d_in[15];
  const float* sgb   = (const float*)d_in[16];
  const float* fcW   = (const float*)d_in[17];
  const float* fcb   = (const float*)d_in[18];
  float* out = (float*)d_out;

  char* ws = (char*)d_ws;
  size_t off = 0;
  auto take = [&](size_t bytes) -> char* {
    char* p = ws + off;
    off = (off + bytes + 255) & ~(size_t)255;
    return p;
  };
  const int K0 = ZDIM + HDIM;      // 1152
  const int K1 = 2 * HDIM;         // 2048
  const int KA = ENCD + HDIM;      // 1536
  unsigned short* W0c  = (unsigned short*)take((size_t)H4 * K0 * 2);
  unsigned short* W1c  = (unsigned short*)take((size_t)H4 * K1 * 2);
  unsigned short* WinB = (unsigned short*)take((size_t)ENCD * HDIM * 2);
  unsigned short* WoutB= (unsigned short*)take((size_t)HDIM * KA * 2);
  unsigned short* WmB  = (unsigned short*)take((size_t)ZDIM * HDIM * 2);
  unsigned short* WsB  = (unsigned short*)take((size_t)ZDIM * HDIM * 2);
  unsigned short* WfcB = (unsigned short*)take((size_t)ENCD * ZDIM * 2);
  float* b0s  = (float*)take((size_t)H4 * 4);
  float* b1s  = (float*)take((size_t)H4 * 4);
  unsigned short* X0  = (unsigned short*)take((size_t)BSZ * K0 * 2);
  unsigned short* X1  = (unsigned short*)take((size_t)BSZ * K1 * 2);
  unsigned short* XA  = (unsigned short*)take((size_t)BSZ * KA * 2);
  unsigned short* CXT = (unsigned short*)take((size_t)BSZ * HDIM * 2);
  float* GATES = (float*)take((size_t)BSZ * H4 * 4);
  float* Qb    = (float*)take((size_t)BSZ * ENCD * 4);
  float* c0    = (float*)take((size_t)BSZ * HDIM * 4);
  float* c1    = (float*)take((size_t)BSZ * HDIM * 4);
  unsigned short* Zall = (unsigned short*)take((size_t)SEQL * BSZ * ZDIM * 2);

  auto cdiv = [](long a, long b) { return (unsigned)((a + b - 1) / b); };

  // --- pack weights to bf16 (concatenated [Wih|Whh] for single K-reduction GEMMs) ---
  k_pack_bf16<<<cdiv((long)H4 * ZDIM, 256), 256, 0, stream>>>(Wih0, W0c, H4, ZDIM, K0);
  k_pack_bf16<<<cdiv((long)H4 * HDIM, 256), 256, 0, stream>>>(Whh0, W0c + ZDIM, H4, HDIM, K0);
  k_pack_bf16<<<cdiv((long)H4 * HDIM, 256), 256, 0, stream>>>(Wih1, W1c, H4, HDIM, K1);
  k_pack_bf16<<<cdiv((long)H4 * HDIM, 256), 256, 0, stream>>>(Whh1, W1c + HDIM, H4, HDIM, K1);
  k_pack_bf16<<<cdiv((long)ENCD * HDIM, 256), 256, 0, stream>>>(Win, WinB, ENCD, HDIM, HDIM);
  k_pack_bf16<<<cdiv((long)HDIM * KA, 256), 256, 0, stream>>>(Wout, WoutB, HDIM, KA, KA);
  k_pack_bf16<<<cdiv((long)ZDIM * HDIM, 256), 256, 0, stream>>>(muW, WmB, ZDIM, HDIM, HDIM);
  k_pack_bf16<<<cdiv((long)ZDIM * HDIM, 256), 256, 0, stream>>>(sgW, WsB, ZDIM, HDIM, HDIM);
  k_pack_bf16<<<cdiv((long)ENCD * ZDIM, 256), 256, 0, stream>>>(fcW, WfcB, ENCD, ZDIM, ZDIM);
  k_addf<<<cdiv(H4, 256), 256, 0, stream>>>(bih0, bhh0, b0s, H4);
  k_addf<<<cdiv(H4, 256), 256, 0, stream>>>(bih1, bhh1, b1s, H4);
  k_init_state<<<cdiv((long)BSZ * HDIM, 256), 256, 0, stream>>>(initz, X0, X1, XA, c0, c1);

  // --- sequential decode: 8 kernels per timestep ---
  for (int t = 0; t < SEQL; ++t) {
    // LSTM layer 0: gates = [z|h0] @ W0c^T + b0  (waves = (512/32)*(4096/64) = 1024)
    k_gemm_bias_f32<<<128, 256, 0, stream>>>(X0, K0, W0c, K0, b0s, GATES, H4, BSZ, H4, K0);
    k_lstm_cell<<<cdiv((long)BSZ * HDIM, 256), 256, 0, stream>>>(GATES, c0, X0 + ZDIM, K0, X1, K1);
    // LSTM layer 1: gates = [h0|h1] @ W1c^T + b1
    k_gemm_bias_f32<<<128, 256, 0, stream>>>(X1, K1, W1c, K1, b1s, GATES, H4, BSZ, H4, K1);
    k_lstm_cell<<<cdiv((long)BSZ * HDIM, 256), 256, 0, stream>>>(GATES, c1, X1 + HDIM, K1, XA + ENCD, KA);
    // q = h1 @ Win^T  (waves = 16*8 = 128)
    k_gemm_bias_f32<<<16, 256, 0, stream>>>(X1 + HDIM, K1, WinB, HDIM, nullptr, Qb, ENCD, BSZ, ENCD, HDIM);
    // softmax attention + context -> XA[:, :ENCD] (bf16)
    k_attn<<<BSZ, 128, 0, stream>>>(enc, Qb, XA, KA);
    // cxt = tanh([ctx|h1] @ Wout^T)  (waves = 16*16 = 256)
    k_gemm_tanh_bf16<<<32, 256, 0, stream>>>(XA, KA, WoutB, KA, CXT, HDIM, BSZ, HDIM, KA);
    // mu/sigma GEMMs fused with reparameterized sample -> z (waves = 16*8 = 128)
    k_gemm_musg_sample<<<16, 256, 0, stream>>>(CXT, WmB, WsB, mub, sgb,
                                               eps + (size_t)t * BSZ * ZDIM,
                                               X0, Zall + (size_t)t * BSZ * ZDIM);
  }
  // out[b][t][e] = Zall[t*BSZ+b] @ fcW^T + fcb  (waves = 1024*8 = 8192)
  k_gemm_fc<<<1024, 256, 0, stream>>>(Zall, WfcB, fcb, out);
}
